// DotProductAttention_28140625723839
// MI455X (gfx1250) — matrix operand order repaired and checked by reading, not compiled, not run on hardware
//
#include <hip/hip_runtime.h>
#include <hip/hip_bf16.h>
#include <math.h>

typedef __attribute__((ext_vector_type(16))) _Float16 v16h;
typedef __attribute__((ext_vector_type(8)))  _Float16 h8;
typedef __attribute__((ext_vector_type(8)))  float    v8f;
typedef __attribute__((ext_vector_type(4)))  float    f4;

#define B_    2
#define H_    16
#define S_    2048
#define D_    64
#define KT    32           // keys per tile
#define QT    64           // q rows per block (4 waves x 16)
#define NEG_BIG (-1e9f)
#define SCALE 0.03125f     // 1/sqrt(1024)

__device__ __forceinline__ v16h cat16(h8 lo, h8 hi) {
  v16h r;
#pragma unroll
  for (int i = 0; i < 8; ++i) { r[i] = lo[i]; r[8 + i] = hi[i]; }
  return r;
}

// convert 16+16 staged f32 values to f16 and store into K/V LDS buffers
__device__ __forceinline__ void stage_to_lds(int key0, int d0,
    f4 k0, f4 k1, f4 k2, f4 k3, f4 v0, f4 v1, f4 v2, f4 v3,
    _Float16 (*Kl)[D_], _Float16 (*Vl)[KT]) {
  h8 klo, khi;
#pragma unroll
  for (int i = 0; i < 4; ++i) {
    klo[i]     = (_Float16)k0[i];  klo[4 + i] = (_Float16)k1[i];
    khi[i]     = (_Float16)k2[i];  khi[4 + i] = (_Float16)k3[i];
  }
  *(h8*)&Kl[key0][d0]     = klo;
  *(h8*)&Kl[key0][d0 + 8] = khi;
#pragma unroll
  for (int i = 0; i < 4; ++i) {
    Vl[d0 + i][key0]      = (_Float16)v0[i];
    Vl[d0 + 4 + i][key0]  = (_Float16)v1[i];
    Vl[d0 + 8 + i][key0]  = (_Float16)v2[i];
    Vl[d0 + 12 + i][key0] = (_Float16)v3[i];
  }
}

__global__ __launch_bounds__(128)
void fa_wmma_kernel(const float* __restrict__ Q,
                    const float* __restrict__ K,
                    const float* __restrict__ V,
                    const int*   __restrict__ Msk,
                    float* __restrict__ Out) {
  __shared__ alignas(16) _Float16 Klds[2][KT][D_];   // double-buffered [key][d]
  __shared__ alignas(16) _Float16 Vt[2][D_][KT];     // double-buffered [d][key]
  __shared__ alignas(16) _Float16 Pt[4][16][KT];     // per-wave P^T as [q][key]
  __shared__ alignas(16) float    madd[2][KT];

  const int tid  = threadIdx.x;
  const int wid  = tid >> 5;
  const int lane = tid & 31;
  const int hi   = lane >> 4;       // half-wave select
  const int l15  = lane & 15;
  const int h8b  = 8 * hi;

  const int b  = blockIdx.z;
  const int h  = blockIdx.y;
  const int qb = blockIdx.x * QT;

  const size_t headOff = ((size_t)b * H_ + h) * (size_t)S_ * D_;
  const int    qrow    = qb + wid * 16 + l15;   // B-layout: lane holds column n=l15
  const float* qp      = Q + headOff + (size_t)qrow * D_;

  // ---- Q^T B-operands, held in registers across the key loop ----
  // B 32x16 f16 layout: element e -> k(d) = dc*32 + hi*16 + e (16 contiguous d)
  v16h bq[2];
#pragma unroll
  for (int dc = 0; dc < 2; ++dc) {
    const int base = dc * 32 + hi * 16;
    f4 x0 = *(const f4*)(qp + base);
    f4 x1 = *(const f4*)(qp + base + 4);
    f4 x2 = *(const f4*)(qp + base + 8);
    f4 x3 = *(const f4*)(qp + base + 12);
#pragma unroll
    for (int i = 0; i < 4; ++i) {
      bq[dc][i]      = (_Float16)x0[i];
      bq[dc][4 + i]  = (_Float16)x1[i];
      bq[dc][8 + i]  = (_Float16)x2[i];
      bq[dc][12 + i] = (_Float16)x3[i];
    }
  }

  // ---- flash state: O^T accumulators; per-lane scalar row stats (q = l15) ----
  v8f   Oacc[4];
#pragma unroll
  for (int nt = 0; nt < 4; ++nt)
#pragma unroll
    for (int r = 0; r < 8; ++r) Oacc[nt][r] = 0.0f;
  float m_run = -1e30f, l_run = 0.0f;

  const int key0 = tid >> 2;       // 0..31 : staging key
  const int d0   = (tid & 3) * 16; // staging 16-d segment
  const float* kbase = K + headOff + (size_t)key0 * D_ + d0;
  const float* vbase = V + headOff + (size_t)key0 * D_ + d0;

  // ---- prologue: stage tile 0 into buffer 0 ----
  {
    f4 k0 = *(const f4*)(kbase);      f4 k1 = *(const f4*)(kbase + 4);
    f4 k2 = *(const f4*)(kbase + 8);  f4 k3 = *(const f4*)(kbase + 12);
    f4 v0 = *(const f4*)(vbase);      f4 v1 = *(const f4*)(vbase + 4);
    f4 v2 = *(const f4*)(vbase + 8);  f4 v3 = *(const f4*)(vbase + 12);
    stage_to_lds(key0, d0, k0, k1, k2, k3, v0, v1, v2, v3, Klds[0], Vt[0]);
    if (tid < KT) madd[0][tid] = Msk[(size_t)b * S_ + tid] ? 0.0f : NEG_BIG;
  }
  __syncthreads();

  int ib = 0;
  for (int kb = 0; kb < S_; kb += KT, ib ^= 1) {
    const bool more = (kb + KT) < S_;

    // ---- issue next tile's global loads EARLY (outstanding across compute) ----
    f4 nk0, nk1, nk2, nk3, nv0, nv1, nv2, nv3;
    int nm = 0;
    if (more) {
      const float* kp = kbase + (size_t)(kb + KT) * D_;
      const float* vp = vbase + (size_t)(kb + KT) * D_;
      nk0 = *(const f4*)(kp);      nk1 = *(const f4*)(kp + 4);
      nk2 = *(const f4*)(kp + 8);  nk3 = *(const f4*)(kp + 12);
      nv0 = *(const f4*)(vp);      nv1 = *(const f4*)(vp + 4);
      nv2 = *(const f4*)(vp + 8);  nv3 = *(const f4*)(vp + 12);
      if (tid < KT) nm = Msk[(size_t)b * S_ + kb + KT + tid];
      if (kb + 2 * KT < S_) {      // prefetch two tiles ahead
        __builtin_prefetch(kp + (size_t)KT * D_, 0, 1);
        __builtin_prefetch(vp + (size_t)KT * D_, 0, 1);
      }
    }

    // mask addends for this lane's 16 keys (keys r+8*hi and 16+r+8*hi)
    f4 ma0 = *(const f4*)&madd[ib][h8b];
    f4 ma1 = *(const f4*)&madd[ib][h8b + 4];
    f4 ma2 = *(const f4*)&madd[ib][16 + h8b];
    f4 ma3 = *(const f4*)&madd[ib][20 + h8b];

    // ---- S^T = K @ Q^T : key chunks 0..15 (s0), 16..31 (s1); d in 2 chunks ----
    v8f s0, s1;
#pragma unroll
    for (int r = 0; r < 8; ++r) { s0[r] = 0.0f; s1[r] = 0.0f; }
#pragma unroll
    for (int dc = 0; dc < 2; ++dc) {
      const int db = dc * 32 + hi * 16;  // A-layout: d = db + e (16 contiguous)
      h8 a0lo = *(const h8*)&Klds[ib][l15][db];
      h8 a0hi = *(const h8*)&Klds[ib][l15][db + 8];
      s0 = __builtin_amdgcn_wmma_f32_16x16x32_f16(false, cat16(a0lo, a0hi), false,
                                                  bq[dc], (short)0, s0, false, false);
      h8 a1lo = *(const h8*)&Klds[ib][16 + l15][db];
      h8 a1hi = *(const h8*)&Klds[ib][16 + l15][db + 8];
      s1 = __builtin_amdgcn_wmma_f32_16x16x32_f16(false, cat16(a1lo, a1hi), false,
                                                  bq[dc], (short)0, s1, false, false);
    }

    // ---- online softmax: column q=l15 is per-lane; keys live in registers ----
    float p0[8], p1[8];
    p0[0] = s0[0] * SCALE + ma0[0]; p0[1] = s0[1] * SCALE + ma0[1];
    p0[2] = s0[2] * SCALE + ma0[2]; p0[3] = s0[3] * SCALE + ma0[3];
    p0[4] = s0[4] * SCALE + ma1[0]; p0[5] = s0[5] * SCALE + ma1[1];
    p0[6] = s0[6] * SCALE + ma1[2]; p0[7] = s0[7] * SCALE + ma1[3];
    p1[0] = s1[0] * SCALE + ma2[0]; p1[1] = s1[1] * SCALE + ma2[1];
    p1[2] = s1[2] * SCALE + ma2[2]; p1[3] = s1[3] * SCALE + ma2[3];
    p1[4] = s1[4] * SCALE + ma3[0]; p1[5] = s1[5] * SCALE + ma3[1];
    p1[6] = s1[6] * SCALE + ma3[2]; p1[7] = s1[7] * SCALE + ma3[3];

    float t = fmaxf(p0[0], p1[0]);
#pragma unroll
    for (int r = 1; r < 8; ++r) t = fmaxf(t, fmaxf(p0[r], p1[r]));
    t = fmaxf(t, __shfl_xor(t, 16, 32));          // merge the two half-waves

    const float mn    = fmaxf(m_run, t);
    const float alpha = __expf(m_run - mn);
    m_run = mn;

    float rs = 0.0f;
#pragma unroll
    for (int r = 0; r < 8; ++r) {
      p0[r] = __expf(p0[r] - mn);
      p1[r] = __expf(p1[r] - mn);
      rs += p0[r] + p1[r];
    }
    rs += __shfl_xor(rs, 16, 32);                 // total over all 32 keys
    l_run = l_run * alpha + rs;

#pragma unroll
    for (int nt = 0; nt < 4; ++nt)
#pragma unroll
      for (int r = 0; r < 8; ++r) Oacc[nt][r] *= alpha;

    // ---- P^T -> LDS as [q][key]: keys r+8*hi contiguous -> 2x b128 store ----
    h8 pk0, pk1;
#pragma unroll
    for (int i = 0; i < 8; ++i) { pk0[i] = (_Float16)p0[i]; pk1[i] = (_Float16)p1[i]; }
    *(h8*)&Pt[wid][l15][h8b]      = pk0;
    *(h8*)&Pt[wid][l15][16 + h8b] = pk1;

    // B-layout reload: key = 8*hi + (e<8?e:16+e-8)
    h8 plo = *(const h8*)&Pt[wid][l15][h8b];
    h8 phi = *(const h8*)&Pt[wid][l15][16 + h8b];
    v16h bp = cat16(plo, phi);

    // ---- O^T += V^T @ P^T : 4 d-chunks of 16, K=32 key contraction each ----
#pragma unroll
    for (int nt = 0; nt < 4; ++nt) {
      const int d = nt * 16 + l15;   // A-layout: m=l15, key = 8*hi + (e<8?e:16+e-8)
      h8 vlo = *(const h8*)&Vt[ib][d][h8b];
      h8 vhi = *(const h8*)&Vt[ib][d][h8b + 16];
      Oacc[nt] = __builtin_amdgcn_wmma_f32_16x16x32_f16(false, cat16(vlo, vhi), false,
                                                        bp, (short)0, Oacc[nt], false, false);
    }

    // ---- convert + store next tile into the other buffer ----
    if (more) {
      stage_to_lds(key0, d0, nk0, nk1, nk2, nk3, nv0, nv1, nv2, nv3,
                   Klds[ib ^ 1], Vt[ib ^ 1]);
      if (tid < KT) madd[ib ^ 1][tid] = nm ? 0.0f : NEG_BIG;
    }
    __syncthreads();   // single barrier per iteration (2 buffers make this safe)
  }

  // ---- normalize, write out: lane holds 8 consecutive d per chunk -> b128s ----
  const float inv = 1.0f / l_run;
  float* outp = Out + headOff + (size_t)qrow * D_;
#pragma unroll
  for (int nt = 0; nt < 4; ++nt) {
    f4 lo, hv;
#pragma unroll
    for (int i = 0; i < 4; ++i) {
      lo[i] = Oacc[nt][i] * inv;
      hv[i] = Oacc[nt][4 + i] * inv;
    }
    *(f4*)(outp + nt * 16 + h8b)     = lo;   // d = nt*16 + 8*hi + 0..3
    *(f4*)(outp + nt * 16 + h8b + 4) = hv;   // d = nt*16 + 8*hi + 4..7
  }
}

extern "C" void kernel_launch(void* const* d_in, const int* in_sizes, int n_in,
                              void* d_out, int out_size, void* d_ws, size_t ws_size,
                              hipStream_t stream) {
  (void)in_sizes; (void)n_in; (void)out_size; (void)d_ws; (void)ws_size;
  const float* Q   = (const float*)d_in[0];
  const float* K   = (const float*)d_in[1];
  const float* V   = (const float*)d_in[2];
  const int*   Msk = (const int*)d_in[3];
  float* Out = (float*)d_out;

  dim3 grid(S_ / QT, H_, B_);   // (32, 16, 2)
  dim3 block(128);              // 4 waves x 16 q-rows
  fa_wmma_kernel<<<grid, block, 0, stream>>>(Q, K, V, Msk, Out);
}
